// QM9Encoder_25563645345821
// MI455X (gfx1250) — matrix operand order repaired and checked by reading, not compiled, not run on hardware
//
#include <hip/hip_runtime.h>

typedef __attribute__((ext_vector_type(16))) __bf16 v16bf;
typedef __attribute__((ext_vector_type(8)))  float  v8f;

constexpr int kN = 10000, kE = 40000, kD = 64, kB = 512;
constexpr int kFIN = 11, kEF = 5, kHN = 128, kSTEPS = 3;
constexpr int kCHUNK = 4096;              // edges per ew tile (32 MB bf16, L2-resident)

__device__ __forceinline__ float sigmoidf_(float x){ return 1.f/(1.f+__expf(-x)); }

__device__ __forceinline__ void atomicAddF(float* p, float v){
  __hip_atomic_fetch_add(p, v, __ATOMIC_RELAXED, __HIP_MEMORY_SCOPE_AGENT);
}
__device__ __forceinline__ void atomicMaxF(float* p, float v){
  // sign-split trick; storage initialized to -inf (0xFF800000)
  if (v >= 0.f) atomicMax((int*)p, __float_as_int(v));
  else          atomicMin((unsigned int*)p, (unsigned int)__float_as_int(v));
}

// ---------------------------------------------------------------- utilities
__global__ void fill_f32(float* __restrict__ p, float v, int n){
  int i = blockIdx.x*blockDim.x + threadIdx.x;
  if (i < n) p[i] = v;
}

__global__ void deg_count(const int* __restrict__ dst, float* __restrict__ deg){
  int e = blockIdx.x*blockDim.x + threadIdx.x;
  if (e < kE) atomicAddF(&deg[dst[e]], 1.f);
}

// out = relu(x @ lin0_w + lin0_b)   [N,11]@[11,64]
__global__ void lin0_relu(const float* __restrict__ x, const float* __restrict__ w,
                          const float* __restrict__ b, float* __restrict__ h){
  int idx = blockIdx.x*blockDim.x + threadIdx.x;
  if (idx >= kN*kD) return;
  int n = idx >> 6, o = idx & 63;
  float s = b[o];
  #pragma unroll
  for (int i = 0; i < kFIN; ++i) s += x[n*kFIN + i] * w[i*kD + o];
  h[idx] = s > 0.f ? s : 0.f;
}

// a = relu(edge_attr @ net_w1 + net_b1), stored bf16   [E,5]@[5,128]
__global__ void edge_mlp1(const float* __restrict__ ea, const float* __restrict__ w1,
                          const float* __restrict__ b1, __bf16* __restrict__ abf){
  int idx = blockIdx.x*blockDim.x + threadIdx.x;
  if (idx >= kE*kHN) return;
  int e = idx >> 7, j = idx & 127;
  float s = b1[j];
  #pragma unroll
  for (int i = 0; i < kEF; ++i) s += ea[e*kEF + i] * w1[i*kHN + j];
  abf[idx] = (__bf16)(s > 0.f ? s : 0.f);
}

// Pre-swizzle W2 [128,4096] into WMMA B-operand order:
// [kstep 0..3][ncol16 0..255][lane 0..31][slot 0..15]
//   K = kstep*32 + (slot<8 ? 0 : 8) + (lane<16 ? 0 : 8) + slot,  col = ncol16*16 + (lane&15)
__global__ void w2_swizzle(const float* __restrict__ w2, __bf16* __restrict__ w2b){
  int idx = blockIdx.x*blockDim.x + threadIdx.x;
  if (idx >= 4*256*32*16) return;
  int slot =  idx        & 15;
  int lane = (idx >> 4)  & 31;
  int nc16 = (idx >> 9)  & 255;
  int ks   =  idx >> 17;
  int K = ks*32 + ((slot < 8) ? 0 : 8) + ((lane < 16) ? 0 : 8) + slot;
  int c = nc16*16 + (lane & 15);
  w2b[idx] = (__bf16)w2[K*4096 + c];
}

// ew_chunk[m, c] = sum_k a[e0+m, k]*W2[k, c] + b2[c]   (bf16 out, f32 accum)
// one wave per block; wave computes a 64x64 output block: 4 M-tiles x 4 N-tiles,
// 64 WMMAs. All inner-loop addresses are immediate offsets off two per-lane
// base pointers; __launch_bounds__(32,1) gives the full VGPR budget (~190 used).
__global__ void __launch_bounds__(32, 1)
ew_gemm(const __bf16* __restrict__ abf, const __bf16* __restrict__ w2b,
        const float* __restrict__ b2, __bf16* __restrict__ ew, int e0){
  const int lane   = threadIdx.x;           // 0..31
  const int nstrip = blockIdx.x;            // 0..63  -> 64 columns
  const int m0     = blockIdx.y * 64;       // chunk-local edge block (64 edges)
  const int lrow   = lane & 15;
  const int hiHalf = lane >> 4;

  // A base: row (m0+lrow), K-half offset for lanes>=16 (16-bit A 16x32 layout).
  // Tile offsets: mt*16 rows = mt*2048 elem (4096B), ks*32 elem (64B) - immediates.
  const __bf16* a0 = abf + (size_t)(e0 + m0 + lrow)*kHN + (hiHalf ? 8 : 0);
  // B base: per-lane 32B slice; nt stride 512 elem (1024B), ks stride 131072 elem
  // (262144B) - all within signed 24-bit instruction offsets.
  const __bf16* bp = w2b + (size_t)nstrip*2048 + (size_t)lane*16;

  v8f acc[4][4] = {};   // [mt][nt]
  #pragma unroll
  for (int ks = 0; ks < 4; ++ks){
    union U { int4 q[2]; v16bf v; };
    U A[4];
    #pragma unroll
    for (int mt = 0; mt < 4; ++mt){
      const int4* pa = (const int4*)(a0 + mt*2048 + ks*32);
      A[mt].q[0] = pa[0];          // K offsets 0..7   (16B)
      A[mt].q[1] = pa[2];          // K offsets 16..23 (16B)
    }
    #pragma unroll
    for (int nt = 0; nt < 4; ++nt){
      U Bt;
      const int4* pb = (const int4*)(bp + ks*131072 + nt*512);
      Bt.q[0] = pb[0]; Bt.q[1] = pb[1];    // 32 contiguous bytes per lane
      #pragma unroll
      for (int mt = 0; mt < 4; ++mt)
        acc[mt][nt] = __builtin_amdgcn_wmma_f32_16x16x32_bf16(
            false, A[mt].v, false, Bt.v, (short)0, acc[mt][nt], false, false);
    }
  }
  // C/D layout: VGPR r -> M = r (lanes 0-15) / r+8 (lanes 16-31); lane&15 -> column
  const int mofs = hiHalf ? 8 : 0;
  __bf16* erow = ew + (size_t)(m0 + mofs)*4096 + nstrip*64 + lrow;
  #pragma unroll
  for (int nt = 0; nt < 4; ++nt){
    float bias = b2[nstrip*64 + nt*16 + lrow];
    #pragma unroll
    for (int mt = 0; mt < 4; ++mt){
      #pragma unroll
      for (int r = 0; r < 8; ++r)
        erow[(size_t)(mt*16 + r)*4096 + nt*16] = (__bf16)(acc[mt][nt][r] + bias);
    }
  }
}

// msg[e,o] = sum_i h[src[e],i] * ew[e, i*64+o];  atomically scatter into agg[dst]
__global__ void msg_matvec(const __bf16* __restrict__ ew, const int* __restrict__ src,
                           const int* __restrict__ dst, const float* __restrict__ h,
                           float* __restrict__ agg, int e0){
  const int eloc = blockIdx.x;
  const int o    = threadIdx.x;            // 64 threads = one edge
  const int eg   = e0 + eloc;
  __shared__ float hs[kD];
  hs[o] = h[(size_t)src[eg]*kD + o];
  __syncthreads();
  const __bf16* row = ew + (size_t)eloc*4096 + o;
  float acc = 0.f;
  #pragma unroll 8
  for (int i = 0; i < kD; ++i) acc += hs[i] * (float)row[i*kD];
  atomicAddF(&agg[(size_t)dst[eg]*kD + o], acc);
}

// m = relu(agg/deg + h@conv_root + bias); GRU(h, m) in place (per-node local)
__global__ void node_update(float* __restrict__ h, const float* __restrict__ agg,
                            const float* __restrict__ deg,
                            const float* __restrict__ croot, const float* __restrict__ cbias,
                            const float* __restrict__ gwih, const float* __restrict__ gwhh,
                            const float* __restrict__ gbih, const float* __restrict__ gbhh){
  const int n = blockIdx.x, o = threadIdx.x;  // 64 threads per node
  __shared__ float hrow[kD], mrow[kD];
  hrow[o] = h[(size_t)n*kD + o];
  __syncthreads();
  float s = cbias[o];
  #pragma unroll 8
  for (int i = 0; i < kD; ++i) s += hrow[i] * croot[i*kD + o];
  float d = deg[n]; d = d < 1.f ? 1.f : d;
  float m = agg[(size_t)n*kD + o] / d + s;
  mrow[o] = m > 0.f ? m : 0.f;
  __syncthreads();
  float gr = gbih[o]        + gbhh[o];
  float gz = gbih[kD + o];
  float gzh = gbhh[kD + o];
  float gn = gbih[2*kD + o];
  float hn = gbhh[2*kD + o];
  float grh = 0.f;
  #pragma unroll 4
  for (int i = 0; i < kD; ++i){
    float mi = mrow[i], hi = hrow[i];
    gr  += mi * gwih[i*192 + o];
    grh += hi * gwhh[i*192 + o];
    gz  += mi * gwih[i*192 + kD + o];
    gzh += hi * gwhh[i*192 + kD + o];
    gn  += mi * gwih[i*192 + 2*kD + o];
    hn  += hi * gwhh[i*192 + 2*kD + o];
  }
  float r  = sigmoidf_(gr + grh);
  float z  = sigmoidf_(gz + gzh);
  float ng = tanhf(gn + r*hn);
  h[(size_t)n*kD + o] = (1.f - z)*ng + z*hrow[o];
}

// ---------------------------------------------------------------- Set2Set
__global__ void lstm_cell(const float* __restrict__ qstar, float* __restrict__ hl,
                          float* __restrict__ cl,
                          const float* __restrict__ wih, const float* __restrict__ whh,
                          const float* __restrict__ bih, const float* __restrict__ bhh){
  const int b = blockIdx.x, o = threadIdx.x;  // 64 threads per graph
  __shared__ float q[2*kD], hh[kD];
  q[o] = qstar[(size_t)b*2*kD + o];
  q[kD + o] = qstar[(size_t)b*2*kD + kD + o];
  hh[o] = hl[(size_t)b*kD + o];
  __syncthreads();
  float gi = bih[o]        + bhh[o];
  float gf = bih[kD+o]     + bhh[kD+o];
  float gg = bih[2*kD+o]   + bhh[2*kD+o];
  float go = bih[3*kD+o]   + bhh[3*kD+o];
  #pragma unroll 4
  for (int i = 0; i < 2*kD; ++i){
    float qi = q[i];
    gi += qi*wih[i*256 + o];       gf += qi*wih[i*256 + kD + o];
    gg += qi*wih[i*256 + 2*kD+o];  go += qi*wih[i*256 + 3*kD + o];
  }
  #pragma unroll 4
  for (int i = 0; i < kD; ++i){
    float hi = hh[i];
    gi += hi*whh[i*256 + o];       gf += hi*whh[i*256 + kD + o];
    gg += hi*whh[i*256 + 2*kD+o];  go += hi*whh[i*256 + 3*kD + o];
  }
  float c = sigmoidf_(gf)*cl[(size_t)b*kD + o] + sigmoidf_(gi)*tanhf(gg);
  float hv = sigmoidf_(go)*tanhf(c);
  cl[(size_t)b*kD + o] = c;
  hl[(size_t)b*kD + o] = hv;
}

__global__ void att_dot(const float* __restrict__ h, const float* __restrict__ hl,
                        const int* __restrict__ batch, float* __restrict__ evec){
  int n = blockIdx.x*blockDim.x + threadIdx.x;
  if (n >= kN) return;
  const float* hr = h  + (size_t)n*kD;
  const float* qr = hl + (size_t)batch[n]*kD;
  float s = 0.f;
  #pragma unroll 8
  for (int i = 0; i < kD; ++i) s += hr[i]*qr[i];
  evec[n] = s;
}

__global__ void att_max(const float* __restrict__ evec, const int* __restrict__ batch,
                        float* __restrict__ emax){
  int n = blockIdx.x*blockDim.x + threadIdx.x;
  if (n < kN) atomicMaxF(&emax[batch[n]], evec[n]);
}

__global__ void att_expsum(const float* __restrict__ evec, const int* __restrict__ batch,
                           const float* __restrict__ emax, float* __restrict__ ee,
                           float* __restrict__ esum){
  int n = blockIdx.x*blockDim.x + threadIdx.x;
  if (n >= kN) return;
  int b = batch[n];
  float v = __expf(evec[n] - emax[b]);
  ee[n] = v;
  atomicAddF(&esum[b], v);
}

__global__ void att_scatter(const float* __restrict__ h, const float* __restrict__ ee,
                            const float* __restrict__ esum, const int* __restrict__ batch,
                            float* __restrict__ rvec){
  int idx = blockIdx.x*blockDim.x + threadIdx.x;
  if (idx >= kN*kD) return;
  int n = idx >> 6, o = idx & 63;
  int b = batch[n];
  float a = ee[n] / esum[b];
  atomicAddF(&rvec[(size_t)b*kD + o], a * h[idx]);
}

__global__ void qstar_build(const float* __restrict__ hl, const float* __restrict__ rvec,
                            float* __restrict__ qstar){
  int idx = blockIdx.x*blockDim.x + threadIdx.x;
  if (idx >= kB*kD) return;
  int b = idx >> 6, o = idx & 63;
  qstar[(size_t)b*2*kD + o]      = hl[idx];
  qstar[(size_t)b*2*kD + kD + o] = rvec[idx];
}

__global__ void lin1_relu(const float* __restrict__ q, const float* __restrict__ w,
                          const float* __restrict__ b, float* __restrict__ out){
  int idx = blockIdx.x*blockDim.x + threadIdx.x;
  if (idx >= kB*kD) return;
  int bb = idx >> 6, o = idx & 63;
  float s = b[o];
  #pragma unroll 4
  for (int i = 0; i < 2*kD; ++i) s += q[(size_t)bb*2*kD + i] * w[i*kD + o];
  out[idx] = s > 0.f ? s : 0.f;
}

// ---------------------------------------------------------------- host
extern "C" void kernel_launch(void* const* d_in, const int* in_sizes, int n_in,
                              void* d_out, int out_size, void* d_ws, size_t ws_size,
                              hipStream_t stream){
  (void)in_sizes; (void)n_in; (void)out_size; (void)ws_size;
  const float* x         = (const float*)d_in[0];
  const float* edge_attr = (const float*)d_in[1];
  const int*   edge_idx  = (const int*)  d_in[2];
  const int*   batch     = (const int*)  d_in[3];
  const float* lin0_w    = (const float*)d_in[4];
  const float* lin0_b    = (const float*)d_in[5];
  const float* net_w1    = (const float*)d_in[6];
  const float* net_b1    = (const float*)d_in[7];
  const float* net_w2    = (const float*)d_in[8];
  const float* net_b2    = (const float*)d_in[9];
  const float* conv_root = (const float*)d_in[10];
  const float* conv_bias = (const float*)d_in[11];
  const float* gru_w_ih  = (const float*)d_in[12];
  const float* gru_w_hh  = (const float*)d_in[13];
  const float* gru_b_ih  = (const float*)d_in[14];
  const float* gru_b_hh  = (const float*)d_in[15];
  const float* lstm_w_ih = (const float*)d_in[16];
  const float* lstm_w_hh = (const float*)d_in[17];
  const float* lstm_b_ih = (const float*)d_in[18];
  const float* lstm_b_hh = (const float*)d_in[19];
  const float* lin1_w    = (const float*)d_in[20];
  const float* lin1_b    = (const float*)d_in[21];

  const int* srcI = edge_idx;        // edge_index[0]
  const int* dstI = edge_idx + kE;   // edge_index[1]

  char* ws = (char*)d_ws;
  size_t off = 0;
  auto alloc = [&](size_t bytes) -> char* {
    char* p = ws + off;
    off = (off + bytes + 255) & ~(size_t)255;
    return p;
  };
  float*  deg   = (float*) alloc((size_t)kN*4);
  float*  h     = (float*) alloc((size_t)kN*kD*4);
  float*  agg   = (float*) alloc((size_t)kN*kD*4);
  __bf16* abf   = (__bf16*)alloc((size_t)kE*kHN*2);
  __bf16* w2b   = (__bf16*)alloc((size_t)kHN*4096*2);
  __bf16* ewc   = (__bf16*)alloc((size_t)kCHUNK*4096*2);
  float*  qstar = (float*) alloc((size_t)kB*256*4);   // qstar(128)|hl(64)|cl(64)
  float*  hl    = qstar + (size_t)kB*2*kD;
  float*  cl    = hl    + (size_t)kB*kD;
  float*  evec  = (float*) alloc((size_t)kN*4);
  float*  ee    = (float*) alloc((size_t)kN*4);
  float*  emax  = (float*) alloc((size_t)kB*4);
  float*  esum  = (float*) alloc((size_t)kB*(1+kD)*4); // esum | rvec contiguous
  float*  rvec  = esum + kB;

  const int T = 256;
  fill_f32 <<<(kN+T-1)/T, T, 0, stream>>>(deg, 0.f, kN);
  deg_count<<<(kE+T-1)/T, T, 0, stream>>>(dstI, deg);
  lin0_relu<<<(kN*kD+T-1)/T, T, 0, stream>>>(x, lin0_w, lin0_b, h);
  edge_mlp1<<<(kE*kHN+T-1)/T, T, 0, stream>>>(edge_attr, net_w1, net_b1, abf);
  w2_swizzle<<<(4*256*32*16+T-1)/T, T, 0, stream>>>(net_w2, w2b);

  for (int step = 0; step < kSTEPS; ++step){
    fill_f32<<<(kN*kD+T-1)/T, T, 0, stream>>>(agg, 0.f, kN*kD);
    for (int e0 = 0; e0 < kE; e0 += kCHUNK){
      int ce = (kE - e0) < kCHUNK ? (kE - e0) : kCHUNK;   // multiples of 64 (4096 / 3136)
      ew_gemm   <<<dim3(64, ce/64), 32, 0, stream>>>(abf, w2b, net_b2, ewc, e0);
      msg_matvec<<<ce, kD, 0, stream>>>(ewc, srcI, dstI, h, agg, e0);
    }
    node_update<<<kN, kD, 0, stream>>>(h, agg, deg, conv_root, conv_bias,
                                       gru_w_ih, gru_w_hh, gru_b_ih, gru_b_hh);
  }

  fill_f32<<<(kB*256+T-1)/T, T, 0, stream>>>(qstar, 0.f, kB*256);
  for (int t = 0; t < kSTEPS; ++t){
    lstm_cell<<<kB, kD, 0, stream>>>(qstar, hl, cl,
                                     lstm_w_ih, lstm_w_hh, lstm_b_ih, lstm_b_hh);
    fill_f32<<<(kB+T-1)/T, T, 0, stream>>>(emax, -__builtin_huge_valf(), kB);
    fill_f32<<<(kB*(1+kD)+T-1)/T, T, 0, stream>>>(esum, 0.f, kB*(1+kD));
    att_dot    <<<(kN+T-1)/T, T, 0, stream>>>(h, hl, batch, evec);
    att_max    <<<(kN+T-1)/T, T, 0, stream>>>(evec, batch, emax);
    att_expsum <<<(kN+T-1)/T, T, 0, stream>>>(evec, batch, emax, ee, esum);
    att_scatter<<<(kN*kD+T-1)/T, T, 0, stream>>>(h, ee, esum, batch, rvec);
    qstar_build<<<(kB*kD+T-1)/T, T, 0, stream>>>(hl, rvec, qstar);
  }
  lin1_relu<<<(kB*kD+T-1)/T, T, 0, stream>>>(qstar, lin1_w, lin1_b, (float*)d_out);
}
